// Encoder_23003844837392
// MI455X (gfx1250) — compile-verified, matrix operand order checked by reading, not run on hardware
//
#include <hip/hip_runtime.h>
#include <hip/hip_bf16.h>
#include <math.h>

// ---------------------------------------------------------------------------
// CDNA5 (gfx1250) implementation: bf16 WMMA GEMMs + flash attention.
// Shapes: B=4, L_IN=1024, D=D_INNER=512, H=8, DK=64, L_TOT=1360, N_LAYER=2.
// ---------------------------------------------------------------------------

typedef __attribute__((ext_vector_type(16))) __bf16 v16bf;
typedef __attribute__((ext_vector_type(8)))  float  v8f;
union FragU { unsigned int u[8]; uint4 q[2]; v16bf v; };

#define NB 4
#define LT 1360
#define DD 512
#define NH 8
#define DKH 64
#define MASKW 48   // 1360 bits -> 43 u32, padded to 48

__device__ __forceinline__ unsigned short f2bf(float f) {
  unsigned int u = __float_as_uint(f);
  unsigned int r = (u + 0x7FFFu + ((u >> 16) & 1u)) >> 16;
  return (unsigned short)r;
}

// ----------------------------- converts ------------------------------------
__global__ void k_cvt_bf16(const float* __restrict__ in, unsigned short* __restrict__ out, size_t n) {
  for (size_t i = blockIdx.x * (size_t)blockDim.x + threadIdx.x; i < n; i += (size_t)gridDim.x * blockDim.x)
    out[i] = f2bf(in[i]);
}

// conv_W (3, O=512, I=512, K=4) -> per level (K*I=2048, O=512) row-major bf16
__global__ void k_cvt_convw(const float* __restrict__ w, unsigned short* __restrict__ out, size_t n) {
  for (size_t i = blockIdx.x * (size_t)blockDim.x + threadIdx.x; i < n; i += (size_t)gridDim.x * blockDim.x) {
    size_t lvl = i / (2048u * 512u);
    size_t r2  = i % (2048u * 512u);
    size_t row = r2 / 512u;          // 0..2047 = k*512 + ii
    size_t o   = r2 % 512u;
    size_t k   = row / 512u;
    size_t ii  = row % 512u;
    out[i] = f2bf(w[((lvl * 512u + o) * 512u + ii) * 4u + k]);
  }
}

// ------------------------------- GEMM ---------------------------------------
// C(M,N) f32 = A(M,K)bf16 * B(K,N)bf16 (+bias) ; transB: B given as (N,K).
// K must be a multiple of 64. If !transB, N must be a multiple of 64.
// epi: 0=none 1=relu 2=scale(1/sqrt(1.00001)) then ELU
__global__ __launch_bounds__(128) void k_gemm_wmma(
    const unsigned short* __restrict__ A, const unsigned short* __restrict__ Bm,
    float* __restrict__ Cf, unsigned short* __restrict__ Cbf,
    const float* __restrict__ bias, int M, int N, int K, int transB, int epi) {
  __shared__ unsigned int Ash[64 * 36];   // 64 rows x 32 dwords (+4 pad), 16B-aligned rows
  __shared__ unsigned int Bsh[64 * 36];
  const int tid  = threadIdx.x;
  const int lane = tid & 31;
  const int w    = tid >> 5;
  const int wr   = w >> 1, wc = w & 1;
  const int m16  = lane & 15, hiL = lane >> 4;
  const int mbase = blockIdx.y * 64;
  const int nbase = blockIdx.x * 64;

  v8f acc[2][2];
  for (int i = 0; i < 2; ++i)
    for (int j = 0; j < 2; ++j)
      for (int e = 0; e < 8; ++e) acc[i][j][e] = 0.0f;

  const int kTiles = K >> 6;
  for (int kt = 0; kt < kTiles; ++kt) {
    const int kk = kt << 6;
    // ---- stage A tile 64x64 bf16: wide b128 loads, clamp+select (no branches)
    {
      const int r  = tid >> 1;
      const int hc = (tid & 1) * 16;      // dword column (16 dwords per thread)
      const int gm = mbase + r;
      const bool ok = gm < M;
      const int gmc = ok ? gm : (M - 1);
      const unsigned short* srow = A + (size_t)gmc * K + kk + hc * 2;
      if (kt + 1 < kTiles) __builtin_prefetch(srow + 64, 0, 3);  // next K-tile
      const uint4* src = (const uint4*)srow;
      uint4* dst = (uint4*)&Ash[r * 36 + hc];
      #pragma unroll
      for (int i = 0; i < 4; ++i) {
        uint4 v = src[i];
        if (!ok) { v.x = 0u; v.y = 0u; v.z = 0u; v.w = 0u; }
        dst[i] = v;
      }
    }
    if (transB) {
      const int r  = tid >> 1;
      const int hc = (tid & 1) * 16;
      const int gn = nbase + r;
      const bool ok = gn < N;
      const int gnc = ok ? gn : (N - 1);
      const unsigned short* srow = Bm + (size_t)gnc * K + kk + hc * 2;
      if (kt + 1 < kTiles) __builtin_prefetch(srow + 64, 0, 3);
      const uint4* src = (const uint4*)srow;
      uint4* dst = (uint4*)&Bsh[r * 36 + hc];
      #pragma unroll
      for (int i = 0; i < 4; ++i) {
        uint4 v = src[i];
        if (!ok) { v.x = 0u; v.y = 0u; v.z = 0u; v.w = 0u; }
        dst[i] = v;
      }
    } else {
      // B (K,N): stage transposed -> Bsh[n][k]  (N multiple of 64: no guards)
      unsigned short* Bs16 = (unsigned short*)Bsh;
      const int kr = tid >> 1;            // 64 k-rows
      const int nh = (tid & 1) * 32;      // 32 n values (16 dwords)
      const unsigned short* srow = Bm + (size_t)(kk + kr) * N + nbase + nh;
      if (kt + 1 < kTiles) __builtin_prefetch(srow + (size_t)64 * N, 0, 3);
      const unsigned int* src = (const unsigned int*)srow;
      #pragma unroll
      for (int i = 0; i < 16; ++i) {
        const unsigned int vv = src[i];
        const int n = nh + i * 2;
        Bs16[(n) * 72 + kr]     = (unsigned short)(vv & 0xFFFFu);
        Bs16[(n + 1) * 72 + kr] = (unsigned short)(vv >> 16);
      }
    }
    __syncthreads();

    #pragma unroll
    for (int ks = 0; ks < 2; ++ks) {
      FragU af[2], bfr[2];
      #pragma unroll
      for (int f = 0; f < 2; ++f) {
        const int mr = wr * 32 + f * 16 + m16;
        const int nc = wc * 32 + f * 16 + m16;
        const int ba = mr * 36 + ks * 16 + hiL * 4;
        const int bb = nc * 36 + ks * 16 + hiL * 4;
        af[f].q[0]  = *(const uint4*)&Ash[ba];
        af[f].q[1]  = *(const uint4*)&Ash[ba + 8];
        bfr[f].q[0] = *(const uint4*)&Bsh[bb];
        bfr[f].q[1] = *(const uint4*)&Bsh[bb + 8];
      }
      #pragma unroll
      for (int fr = 0; fr < 2; ++fr)
        #pragma unroll
        for (int fc = 0; fc < 2; ++fc)
          acc[fr][fc] = __builtin_amdgcn_wmma_f32_16x16x32_bf16(
              false, af[fr].v, false, bfr[fc].v, (short)0, acc[fr][fc], false, false);
    }
    __syncthreads();
  }

  const float sc = rsqrtf(1.0f + 1e-5f);
  for (int fr = 0; fr < 2; ++fr)
    for (int fc = 0; fc < 2; ++fc) {
      const int nn = nbase + wc * 32 + fc * 16 + m16;
      if (nn >= N) continue;
      const float bv = bias ? bias[nn] : 0.0f;
      for (int v = 0; v < 8; ++v) {
        const int mm = mbase + wr * 32 + fr * 16 + v + hiL * 8;
        if (mm >= M) continue;
        float x = acc[fr][fc][v] + bv;
        if (epi == 1) x = fmaxf(x, 0.0f);
        else if (epi == 2) { x *= sc; x = (x > 0.0f) ? x : (__expf(x) - 1.0f); }
        const size_t o = (size_t)mm * N + nn;
        if (Cf)  Cf[o]  = x;
        if (Cbf) Cbf[o] = f2bf(x);
      }
    }
}

// --------------------------- temporal mask ----------------------------------
__device__ __forceinline__ void lvl_of(int i, int& l, int& s) {
  if (i < 1024)      { l = 0; s = 0; }
  else if (i < 1280) { l = 1; s = 1024; }
  else if (i < 1344) { l = 2; s = 1280; }
  else               { l = 3; s = 1344; }
}
__device__ __forceinline__ bool t_allowed(int i, int j) {
  int li, si, lj, sj;
  lvl_of(i, li, si); lvl_of(j, lj, sj);
  if (li == lj) { int d = i - j; return (d <= 2) && (d >= -2); }
  if (li == lj + 1) return ((j - sj) >> 2) == (i - si);
  if (lj == li + 1) return ((i - si) >> 2) == (j - sj);
  return false;
}

// --------------------------- flash attention --------------------------------
// One wave handles one (b,h,16-row q tile). smask==nullptr => temporal mask.
__global__ __launch_bounds__(128) void k_flash(
    const unsigned short* __restrict__ Q, const unsigned short* __restrict__ Kb,
    const unsigned short* __restrict__ Vb, unsigned short* __restrict__ O,
    const unsigned int* __restrict__ smask, int nTiles) {
  __shared__ unsigned int Psh[4][16 * 20];   // per-wave P tile, 20-dword row stride
  __shared__ unsigned int Vsh[4][64 * 20];   // per-wave transposed V tile Vt[d][k]
  const int w    = threadIdx.x >> 5;
  const int lane = threadIdx.x & 31;
  const int tile = blockIdx.x * 4 + w;
  if (tile >= nTiles) return;
  const int qt = tile % 85;
  const int bh = tile / 85;
  const int h  = bh % NH;
  const int b  = bh / NH;
  const int qbase = qt * 16;
  const int m16 = lane & 15, hiL = lane >> 4;

  const unsigned short* Qh = Q + (size_t)b * LT * DD + h * DKH;
  const unsigned short* Kh = Kb + (size_t)b * LT * DD + h * DKH;
  const unsigned short* Vh = Vb + (size_t)b * LT * DD + h * DKH;
  const unsigned int* mrow0 = smask ? (smask + (size_t)b * LT * MASKW) : nullptr;

  // Q fragments: two contiguous 16B runs per 32-wide d chunk
  FragU aq[2];
  const int qrow = qbase + m16;
  #pragma unroll
  for (int ch = 0; ch < 2; ++ch) {
    const unsigned short* qp = Qh + (size_t)qrow * DD + ch * 32 + hiL * 8;
    aq[ch].q[0] = *(const uint4*)qp;
    aq[ch].q[1] = *(const uint4*)(qp + 16);
  }

  v8f o[4];
  for (int c = 0; c < 4; ++c) for (int e = 0; e < 8; ++e) o[c][e] = 0.0f;
  float mrun[8], lrun[8];
  for (int v = 0; v < 8; ++v) { mrun[v] = -1e30f; lrun[v] = 0.0f; }

  unsigned int* Pw = Psh[w];
  unsigned short* Pw16 = (unsigned short*)Pw;
  unsigned int* Vw = Vsh[w];
  unsigned short* Vw16 = (unsigned short*)Vw;

  for (int kt = 0; kt < 43; ++kt) {
    const int k0 = kt * 32;

    // ---- load this wave's V rows (one 64-value row per lane, wide b128s)
    uint4 vr[8];
    {
      const int krg = k0 + lane;
      const bool ok = krg < LT;
      const int krc = ok ? krg : (LT - 1);
      const uint4* vp = (const uint4*)(Vh + (size_t)krc * DD);
      #pragma unroll
      for (int i = 0; i < 8; ++i) {
        uint4 t = vp[i];
        if (!ok) { t.x = 0u; t.y = 0u; t.z = 0u; t.w = 0u; }
        vr[i] = t;
      }
    }

    // ---- S = Q K^T for two 16-wide column halves
    v8f s[2];
    #pragma unroll
    for (int half = 0; half < 2; ++half) {
      v8f sh; for (int e = 0; e < 8; ++e) sh[e] = 0.0f;
      const int kc  = k0 + half * 16 + m16;
      const bool inr = kc < LT;
      const int kcc = inr ? kc : (LT - 1);
      #pragma unroll
      for (int ch = 0; ch < 2; ++ch) {
        const unsigned short* kp = Kh + (size_t)kcc * DD + ch * 32 + hiL * 8;
        uint4 q0 = *(const uint4*)kp;
        uint4 q1 = *(const uint4*)(kp + 16);
        if (!inr) { q0.x = q0.y = q0.z = q0.w = 0u; q1.x = q1.y = q1.z = q1.w = 0u; }
        FragU kb; kb.q[0] = q0; kb.q[1] = q1;
        sh = __builtin_amdgcn_wmma_f32_16x16x32_bf16(false, aq[ch].v, false, kb.v,
                                                     (short)0, sh, false, false);
      }
      s[half] = sh;
    }

    // ---- scatter V into LDS transposed: Vt[d][k], row stride 40 u16
    {
      const unsigned int* w32 = (const unsigned int*)vr;
      #pragma unroll
      for (int i = 0; i < 32; ++i) {
        const unsigned int word = w32[i];
        const int d = i * 2;
        Vw16[(d) * 40 + lane]     = (unsigned short)(word & 0xFFFFu);
        Vw16[(d + 1) * 40 + lane] = (unsigned short)(word >> 16);
      }
    }

    // ---- mask + scale + online softmax (mask word per row: kc>>5 == kt)
    #pragma unroll
    for (int v = 0; v < 8; ++v) {
      const int q = qbase + v + hiL * 8;
      unsigned int wq = 0u;
      if (mrow0) wq = mrow0[(size_t)q * MASKW + kt];
      float sv[2];
      #pragma unroll
      for (int half = 0; half < 2; ++half) {
        const int kc = k0 + half * 16 + m16;
        bool allowed = kc < LT;
        if (allowed) {
          if (mrow0) allowed = ((wq >> (half * 16 + m16)) & 1u) == 0u;  // top-k bits MASKED
          else       allowed = t_allowed(q, kc);
        }
        sv[half] = allowed ? (s[half][v] * 0.125f) : -1e30f;
      }
      float loc = fmaxf(sv[0], sv[1]);
      for (int off = 1; off < 16; off <<= 1) loc = fmaxf(loc, __shfl_xor(loc, off, 16));
      const float mnew = fmaxf(mrun[v], loc);
      const float al   = __expf(mrun[v] - mnew);
      const float p0 = (sv[0] > -1e29f) ? __expf(sv[0] - mnew) : 0.0f;
      const float p1 = (sv[1] > -1e29f) ? __expf(sv[1] - mnew) : 0.0f;
      float rs = p0 + p1;
      for (int off = 1; off < 16; off <<= 1) rs += __shfl_xor(rs, off, 16);
      lrun[v] = lrun[v] * al + rs;
      mrun[v] = mnew;
      const int m = v + hiL * 8;
      Pw16[m * 40 + m16]      = f2bf(p0);
      Pw16[m * 40 + 16 + m16] = f2bf(p1);
      for (int c = 0; c < 4; ++c) o[c][v] *= al;
    }

    // ---- re-load P as A-fragment (same-wave LDS ops are in-order): two b128s
    FragU ap;
    {
      const int bp = m16 * 20 + hiL * 4;
      ap.q[0] = *(const uint4*)&Pw[bp];
      ap.q[1] = *(const uint4*)&Pw[bp + 8];
    }
    // ---- O += P * V  (V fragments from transposed LDS tile: two b128s each)
    #pragma unroll
    for (int c = 0; c < 4; ++c) {
      FragU vbf;
      const int bb = (c * 16 + m16) * 20 + hiL * 4;
      vbf.q[0] = *(const uint4*)&Vw[bb];
      vbf.q[1] = *(const uint4*)&Vw[bb + 8];
      o[c] = __builtin_amdgcn_wmma_f32_16x16x32_bf16(false, ap.v, false, vbf.v,
                                                     (short)0, o[c], false, false);
    }
  }
  #pragma unroll
  for (int v = 0; v < 8; ++v) {
    const float inv = (lrun[v] > 0.0f) ? (1.0f / lrun[v]) : 0.0f;
    const int q = qbase + v + hiL * 8;
    for (int c = 0; c < 4; ++c)
      O[((size_t)b * LT + q) * DD + h * DKH + c * 16 + m16] = f2bf(o[c][v] * inv);
  }
}

// ------------------------------ row kernels ---------------------------------
__global__ __launch_bounds__(128) void k_ln_rows(
    const float* __restrict__ in, const float* __restrict__ res,
    const float* __restrict__ g, const float* __restrict__ bta, float eps,
    float* __restrict__ outf, unsigned short* __restrict__ outbf, int rows) {
  const int r = blockIdx.x;
  if (r >= rows) return;
  __shared__ float red[128];
  const int t = threadIdx.x;
  const float* xr = in + (size_t)r * DD;
  const float* rr = res ? res + (size_t)r * DD : nullptr;
  float vals[4]; float s = 0.0f;
  for (int i = 0; i < 4; ++i) {
    float v = xr[t + i * 128];
    if (rr) v += rr[t + i * 128];
    vals[i] = v; s += v;
  }
  red[t] = s; __syncthreads();
  for (int off = 64; off; off >>= 1) { if (t < off) red[t] += red[t + off]; __syncthreads(); }
  const float mu = red[0] / (float)DD;
  __syncthreads();
  float s2 = 0.0f;
  for (int i = 0; i < 4; ++i) { const float d = vals[i] - mu; s2 += d * d; }
  red[t] = s2; __syncthreads();
  for (int off = 64; off; off >>= 1) { if (t < off) red[t] += red[t + off]; __syncthreads(); }
  const float inv = rsqrtf(red[0] / (float)DD + eps);
  for (int i = 0; i < 4; ++i) {
    const int idx = t + i * 128;
    const float y = (vals[i] - mu) * inv * g[idx] + bta[idx];
    if (outf)  outf[(size_t)r * DD + idx]  = y;
    if (outbf) outbf[(size_t)r * DD + idx] = f2bf(y);
  }
}

__global__ __launch_bounds__(128) void k_rownorm(
    const float* __restrict__ x, unsigned short* __restrict__ xn, int rows) {
  const int r = blockIdx.x;
  if (r >= rows) return;
  __shared__ float red[128];
  const int t = threadIdx.x;
  float v[4]; float s = 0.0f;
  for (int i = 0; i < 4; ++i) { v[i] = x[(size_t)r * DD + t + i * 128]; s += v[i] * v[i]; }
  red[t] = s; __syncthreads();
  for (int off = 64; off; off >>= 1) { if (t < off) red[t] += red[t + off]; __syncthreads(); }
  const float inv = 1.0f / fmaxf(sqrtf(red[0]), 1e-8f);
  for (int i = 0; i < 4; ++i) xn[(size_t)r * DD + t + i * 128] = f2bf(v[i] * inv);
}

__global__ void k_concat_rows(const float* __restrict__ xenc, const float* __restrict__ pyrup,
                              float* __restrict__ out, size_t n) {
  for (size_t i = blockIdx.x * (size_t)blockDim.x + threadIdx.x; i < n; i += (size_t)gridDim.x * blockDim.x) {
    const size_t c = i % DD, r = i / DD, b = r / LT, j = r % LT;
    out[i] = (j < 1024) ? xenc[((size_t)b * 1024 + j) * DD + c]
                        : pyrup[((size_t)b * 336 + (j - 1024)) * DD + c];
  }
}

__global__ void k_build_pyr(const unsigned short* __restrict__ c1, const unsigned short* __restrict__ c2,
                            const unsigned short* __restrict__ c3, unsigned short* __restrict__ pyr, size_t n) {
  for (size_t i = blockIdx.x * (size_t)blockDim.x + threadIdx.x; i < n; i += (size_t)gridDim.x * blockDim.x) {
    const size_t c = i % DD, r = i / DD, b = r / 336, j = r % 336;
    unsigned short v;
    if (j < 256)      v = c1[((size_t)b * 256 + j) * DD + c];
    else if (j < 320) v = c2[((size_t)b * 64 + (j - 256)) * DD + c];
    else              v = c3[((size_t)b * 16 + (j - 320)) * DD + c];
    pyr[i] = v;
  }
}

__global__ void k_combine3(const float* __restrict__ a, const float* __restrict__ b2,
                           float* __restrict__ x, unsigned short* __restrict__ xbf, size_t n) {
  for (size_t i = blockIdx.x * (size_t)blockDim.x + threadIdx.x; i < n; i += (size_t)gridDim.x * blockDim.x) {
    const float v = (a[i] + b2[i] + x[i]) * (1.0f / 3.0f);
    x[i] = v; xbf[i] = f2bf(v);
  }
}

// Iterative top-k (k<=32) over one sim row per block; sets MASK bits (masked=True).
__global__ __launch_bounds__(256) void k_topk(const float* __restrict__ sim,
                                              unsigned int* __restrict__ mask,
                                              int b, int start, int n, int kcnt) {
  __shared__ float vals[1024];
  __shared__ float bv[256];
  __shared__ int   bi[256];
  const int r = blockIdx.x;
  const int t = threadIdx.x;
  const float* row = sim + (size_t)r * n;
  for (int i = t; i < n; i += 256) vals[i] = row[i];
  __syncthreads();
  unsigned int* mrow = mask + ((size_t)b * LT + start + r) * MASKW;
  for (int it = 0; it < kcnt; ++it) {
    float best = -3e38f; int bidx = n;
    for (int i = t; i < n; i += 256) {
      const float v = vals[i];
      if (v > best || (v == best && i < bidx)) { best = v; bidx = i; }
    }
    bv[t] = best; bi[t] = bidx; __syncthreads();
    for (int off = 128; off; off >>= 1) {
      if (t < off) {
        if (bv[t + off] > bv[t] || (bv[t + off] == bv[t] && bi[t + off] < bi[t])) {
          bv[t] = bv[t + off]; bi[t] = bi[t + off];
        }
      }
      __syncthreads();
    }
    if (t == 0) {
      const int j = bi[0];
      vals[j] = -3e38f;
      const int col = start + j;
      mrow[col >> 5] |= (1u << (col & 31));
    }
    __syncthreads();
  }
}

// ------------------------------- host side ----------------------------------
static inline void gemm(hipStream_t st, const unsigned short* A, const unsigned short* Bm,
                        float* Cf, unsigned short* Cbf, const float* bias,
                        int M, int N, int K, int transB, int epi) {
  dim3 g((N + 63) / 64, (M + 63) / 64);
  k_gemm_wmma<<<g, 128, 0, st>>>(A, Bm, Cf, Cbf, bias, M, N, K, transB, epi);
}

extern "C" void kernel_launch(void* const* d_in, const int* in_sizes, int n_in,
                              void* d_out, int out_size, void* d_ws, size_t ws_size,
                              hipStream_t stream) {
  (void)in_sizes; (void)n_in; (void)out_size; (void)ws_size;
  const float* x_enc   = (const float*)d_in[0];
  const float* down_W  = (const float*)d_in[1];
  const float* down_b  = (const float*)d_in[2];
  const float* conv_W  = (const float*)d_in[3];
  const float* conv_b  = (const float*)d_in[4];
  const float* up_W    = (const float*)d_in[5];
  const float* up_b    = (const float*)d_in[6];
  const float* bc_g    = (const float*)d_in[7];
  const float* bc_b    = (const float*)d_in[8];
  const float* Wq      = (const float*)d_in[9];
  const float* Wk      = (const float*)d_in[10];
  const float* Wv      = (const float*)d_in[11];
  const float* fcW     = (const float*)d_in[12];
  const float* fcb     = (const float*)d_in[13];
  const float* aln_g   = (const float*)d_in[14];
  const float* aln_b   = (const float*)d_in[15];
  const float* fW1     = (const float*)d_in[16];
  const float* fb1     = (const float*)d_in[17];
  const float* fW2     = (const float*)d_in[18];
  const float* fb2     = (const float*)d_in[19];
  const float* fln_g   = (const float*)d_in[20];
  const float* fln_b   = (const float*)d_in[21];

  char* base = (char*)d_ws;
  size_t off = 0;
  auto alloc = [&](size_t bytes) -> void* {
    void* p = base + off;
    off = (off + bytes + 255) & ~(size_t)255;
    return p;
  };
  typedef unsigned short u16;

  u16* wbDown = (u16*)alloc(512 * 512 * 2);
  u16* wbConv = (u16*)alloc((size_t)3 * 2048 * 512 * 2);
  u16* wbUp   = (u16*)alloc(512 * 512 * 2);
  u16* wbQ    = (u16*)alloc((size_t)4 * 512 * 512 * 2);
  u16* wbK    = (u16*)alloc((size_t)4 * 512 * 512 * 2);
  u16* wbV    = (u16*)alloc((size_t)4 * 512 * 512 * 2);
  u16* wbFc   = (u16*)alloc((size_t)4 * 512 * 512 * 2);
  u16* wbF1   = (u16*)alloc((size_t)2 * 512 * 512 * 2);
  u16* wbF2   = (u16*)alloc((size_t)2 * 512 * 512 * 2);
  u16* xencBf = (u16*)alloc((size_t)NB * 1024 * DD * 2);
  u16* t0bf   = (u16*)alloc((size_t)NB * 1024 * DD * 2);
  u16* c1bf   = (u16*)alloc((size_t)NB * 256 * DD * 2);
  u16* c2bf   = (u16*)alloc((size_t)NB * 64 * DD * 2);
  u16* c3bf   = (u16*)alloc((size_t)NB * 16 * DD * 2);
  u16* pyrBf  = (u16*)alloc((size_t)NB * 336 * DD * 2);
  float* pyrupF = (float*)alloc((size_t)NB * 336 * DD * 4);
  float* tmpF = (float*)alloc((size_t)NB * LT * DD * 4);
  float* xF   = (float*)alloc((size_t)NB * LT * DD * 4);
  u16* xBf    = (u16*)alloc((size_t)NB * LT * DD * 2);
  float* toutF = (float*)alloc((size_t)NB * LT * DD * 4);
  float* soutF = (float*)alloc((size_t)NB * LT * DD * 4);
  u16* xnBf   = (u16*)alloc((size_t)NB * LT * DD * 2);
  u16* qBf    = (u16*)alloc((size_t)NB * LT * DD * 2);
  u16* kBf    = (u16*)alloc((size_t)NB * LT * DD * 2);
  u16* vBf    = (u16*)alloc((size_t)NB * LT * DD * 2);
  u16* oBf    = (u16*)alloc((size_t)NB * LT * DD * 2);
  u16* hBf    = (u16*)alloc((size_t)NB * LT * DD * 2);
  float* simF = (float*)alloc((size_t)1024 * 1024 * 4);
  unsigned int* smask = (unsigned int*)alloc((size_t)NB * LT * MASKW * 4);

  const int ROWS = NB * LT;                 // 5440
  const size_t NX = (size_t)ROWS * DD;      // elementwise count
  const int EW_BLOCKS = 1024;

  // ---- weight / input converts (bf16) ----
  k_cvt_bf16<<<512, 256, 0, stream>>>(down_W, wbDown, (size_t)512 * 512);
  k_cvt_convw<<<EW_BLOCKS, 256, 0, stream>>>(conv_W, wbConv, (size_t)3 * 2048 * 512);
  k_cvt_bf16<<<512, 256, 0, stream>>>(up_W, wbUp, (size_t)512 * 512);
  k_cvt_bf16<<<EW_BLOCKS, 256, 0, stream>>>(Wq,  wbQ,  (size_t)4 * 512 * 512);
  k_cvt_bf16<<<EW_BLOCKS, 256, 0, stream>>>(Wk,  wbK,  (size_t)4 * 512 * 512);
  k_cvt_bf16<<<EW_BLOCKS, 256, 0, stream>>>(Wv,  wbV,  (size_t)4 * 512 * 512);
  k_cvt_bf16<<<EW_BLOCKS, 256, 0, stream>>>(fcW, wbFc, (size_t)4 * 512 * 512);
  k_cvt_bf16<<<EW_BLOCKS, 256, 0, stream>>>(fW1, wbF1, (size_t)2 * 512 * 512);
  k_cvt_bf16<<<EW_BLOCKS, 256, 0, stream>>>(fW2, wbF2, (size_t)2 * 512 * 512);
  k_cvt_bf16<<<EW_BLOCKS, 256, 0, stream>>>(x_enc, xencBf, (size_t)NB * 1024 * DD);

  // ---- bottleneck ----
  gemm(stream, xencBf, wbDown, nullptr, t0bf, down_b, NB * 1024, 512, 512, 0, 0);
  gemm(stream, t0bf, wbConv + 0,                 nullptr, c1bf, conv_b + 0,    NB * 256, 512, 2048, 0, 2);
  gemm(stream, c1bf, wbConv + (size_t)2048*512,  nullptr, c2bf, conv_b + 512,  NB * 64,  512, 2048, 0, 2);
  gemm(stream, c2bf, wbConv + (size_t)2*2048*512,nullptr, c3bf, conv_b + 1024, NB * 16,  512, 2048, 0, 2);
  k_build_pyr<<<EW_BLOCKS, 256, 0, stream>>>(c1bf, c2bf, c3bf, pyrBf, (size_t)NB * 336 * DD);
  gemm(stream, pyrBf, wbUp, pyrupF, nullptr, up_b, NB * 336, 512, 512, 0, 0);
  k_concat_rows<<<EW_BLOCKS, 256, 0, stream>>>(x_enc, pyrupF, tmpF, NX);
  k_ln_rows<<<ROWS, 128, 0, stream>>>(tmpF, nullptr, bc_g, bc_b, 1e-5f, xF, xBf, ROWS);

  const int bstart[4] = {0, 1024, 1280, 1344};
  const int bn[4]     = {1024, 256, 64, 16};

  for (int l = 0; l < 2; ++l) {
    // ---- temporal MHA ----
    const size_t wo0 = (size_t)(l * 2 + 0) * 512 * 512;
    gemm(stream, xBf, wbQ + wo0, nullptr, qBf, nullptr, ROWS, 512, 512, 0, 0);
    gemm(stream, xBf, wbK + wo0, nullptr, kBf, nullptr, ROWS, 512, 512, 0, 0);
    gemm(stream, xBf, wbV + wo0, nullptr, vBf, nullptr, ROWS, 512, 512, 0, 0);
    k_flash<<<680, 128, 0, stream>>>(qBf, kBf, vBf, oBf, nullptr, NB * NH * 85);
    gemm(stream, oBf, wbFc + wo0, tmpF, nullptr, fcb + (l * 2 + 0) * 512, ROWS, 512, 512, 0, 0);
    k_ln_rows<<<ROWS, 128, 0, stream>>>(tmpF, xF, aln_g + (size_t)(l * 2 + 0) * 512,
                                        aln_b + (size_t)(l * 2 + 0) * 512, 1e-6f, toutF, nullptr, ROWS);

    // ---- semantic mask ----
    k_rownorm<<<ROWS, 128, 0, stream>>>(xF, xnBf, ROWS);
    hipMemsetAsync(smask, 0, (size_t)NB * LT * MASKW * 4, stream);
    for (int b = 0; b < NB; ++b)
      for (int blk = 0; blk < 4; ++blk) {
        const int st = bstart[blk], n = bn[blk];
        const u16* xn0 = xnBf + ((size_t)b * LT + st) * DD;
        gemm(stream, xn0, xn0, simF, nullptr, nullptr, n, n, 512, 1, 0);
        const int kc = (n < 32) ? n : 32;
        k_topk<<<n, 256, 0, stream>>>(simF, smask, b, st, n, kc);
      }

    // ---- semantic MHA ----
    const size_t wo1 = (size_t)(l * 2 + 1) * 512 * 512;
    gemm(stream, xBf, wbQ + wo1, nullptr, qBf, nullptr, ROWS, 512, 512, 0, 0);
    gemm(stream, xBf, wbK + wo1, nullptr, kBf, nullptr, ROWS, 512, 512, 0, 0);
    gemm(stream, xBf, wbV + wo1, nullptr, vBf, nullptr, ROWS, 512, 512, 0, 0);
    k_flash<<<680, 128, 0, stream>>>(qBf, kBf, vBf, oBf, smask, NB * NH * 85);
    gemm(stream, oBf, wbFc + wo1, tmpF, nullptr, fcb + (l * 2 + 1) * 512, ROWS, 512, 512, 0, 0);
    k_ln_rows<<<ROWS, 128, 0, stream>>>(tmpF, xF, aln_g + (size_t)(l * 2 + 1) * 512,
                                        aln_b + (size_t)(l * 2 + 1) * 512, 1e-6f, soutF, nullptr, ROWS);

    // ---- combine + FFN ----
    k_combine3<<<EW_BLOCKS, 256, 0, stream>>>(toutF, soutF, xF, xBf, NX);
    gemm(stream, xBf, wbF1 + (size_t)l * 512 * 512, nullptr, hBf, fb1 + l * 512, ROWS, 512, 512, 0, 1);
    gemm(stream, hBf, wbF2 + (size_t)l * 512 * 512, tmpF, nullptr, fb2 + l * 512, ROWS, 512, 512, 0, 0);
    float* outF = (l == 1) ? (float*)d_out : xF;
    u16*   outB = (l == 1) ? nullptr : xBf;
    k_ln_rows<<<ROWS, 128, 0, stream>>>(tmpF, xF, fln_g + (size_t)l * 512,
                                        fln_b + (size_t)l * 512, 1e-6f, outF, outB, ROWS);
  }
}
